// RNNFromCell_16200616640717
// MI455X (gfx1250) — matrix operand order repaired and checked by reading, not compiled, not run on hardware
//
#include <hip/hip_runtime.h>
#include <hip/hip_bf16.h>

typedef __attribute__((ext_vector_type(16))) _Float16 v16h;
typedef __attribute__((ext_vector_type(8)))  _Float16 v8h;
typedef __attribute__((ext_vector_type(4)))  _Float16 v4h;
typedef __attribute__((ext_vector_type(8)))  float    v8f;

#define B_TOT   4096
#define T_LEN   80
#define E_DIM   100
#define U_DIM   512
#define EPAD    128
#define BT      32            // batch rows per workgroup
#define NWAVES  8
#define NT_PW   4             // n-tiles (16 cols) per wave: 8*4*16 = 512
#define KT_U    (U_DIM/32)    // 16
#define KT_E    (EPAD/32)     // 4
// padded LDS row strides (in halfs) to avoid 64-bank conflicts on row-strided b128 loads
#define SX_STR  (EPAD + 8)    // 136
#define SH_STR  (U_DIM + 8)   // 520

// ---------------- fast tanh: single V_TANH_F32 trans op on gfx1250 -----------------
__device__ __forceinline__ float fast_tanh(float x)
{
#if defined(__has_builtin)
#if __has_builtin(__builtin_amdgcn_tanhf)
    return __builtin_amdgcn_tanhf(x);
#else
    const float e = __expf(-2.0f * __builtin_fabsf(x));
    const float t = (1.0f - e) / (1.0f + e);
    return __builtin_copysignf(t, x);
#endif
#else
    return tanhf(x);
#endif
}

// ---------------- weight prep: fp32 -> f16, swizzled into B-fragment tile layout ----
// Tile (nt, kt): lane L holds 16 halfs: K = kt*32 + (L>=16 ? 8 : 0) + (e&7) + (e>=8 ? 16 : 0), N = nt*16 + (L&15)
// stored at ws[((nt*KT + kt)*32 + L)*16 + e]  -> per-wave contiguous 1KB tiles.
__global__ void prep_weights(const float* __restrict__ W0, const float* __restrict__ U0,
                             const float* __restrict__ W1, const float* __restrict__ U1,
                             _Float16* __restrict__ ws)
{
    const int NW0 = EPAD * U_DIM;      // 65536
    const int NUU = U_DIM * U_DIM;     // 262144
    const int total = NW0 + 3 * NUU;
    for (int idx = blockIdx.x * blockDim.x + threadIdx.x; idx < total;
         idx += gridDim.x * blockDim.x) {
        const float* src; int Ksrc, KT, local;
        if (idx < NW0)            { src = W0; Ksrc = E_DIM; KT = KT_E; local = idx; }
        else if (idx < NW0+NUU)   { src = U0; Ksrc = U_DIM; KT = KT_U; local = idx - NW0; }
        else if (idx < NW0+2*NUU) { src = W1; Ksrc = U_DIM; KT = KT_U; local = idx - NW0 - NUU; }
        else                      { src = U1; Ksrc = U_DIM; KT = KT_U; local = idx - NW0 - 2*NUU; }
        const int e    = local & 15;
        const int ln   = (local >> 4) & 31;
        const int tile = local >> 9;
        const int kt   = tile % KT;
        const int nt   = tile / KT;
        const int k = kt*32 + ((ln >> 4) << 3) + (e & 7) + ((e >> 3) << 4);
        const int n = nt*16 + (ln & 15);
        const float v = (k < Ksrc) ? src[k * U_DIM + n] : 0.0f;
        ws[idx] = (_Float16)v;
    }
}

// ---------------- A-fragment load from row-major f16 LDS ---------------------------
// A 16x32 f16 layout: lane L holds row M = L&15; elements e: K = kt*32 + (L>>4)*8
// + (e&7) + (e>=8 ? 16 : 0)  => two contiguous 16B chunks at koff and koff+16.
__device__ __forceinline__ v16h load_a(const _Float16* __restrict__ base, int stride,
                                       int lane, int koff32)
{
    const int m    = lane & 15;
    const int koff = koff32 + ((lane >> 4) << 3);
    const _Float16* p = base + m * stride + koff;
    v8h lo = *(const v8h*)(p);
    v8h hi = *(const v8h*)(p + 16);
    v16h r;
#pragma unroll
    for (int i = 0; i < 8; ++i) { r[i] = lo[i]; r[i + 8] = hi[i]; }
    return r;
}

__device__ __forceinline__ v8f wmma16(v16h a, v16h b, v8f c)
{
    return __builtin_amdgcn_wmma_f32_16x16x32_f16(false, a, false, b, (short)0, c,
                                                  false, false);
}

// ---------------- main persistent-state RNN kernel ---------------------------------
__global__ __launch_bounds__(256)
void rnn_main(const int* __restrict__ tokens, const float* __restrict__ emb,
              const _Float16* __restrict__ wsW0, const _Float16* __restrict__ wsU0,
              const _Float16* __restrict__ wsW1, const _Float16* __restrict__ wsU1,
              const float* __restrict__ b0, const float* __restrict__ b1,
              const float* __restrict__ Wout, const float* __restrict__ bout,
              float* __restrict__ out)
{
    __shared__ __align__(16) _Float16 sX [BT * SX_STR];
    __shared__ __align__(16) _Float16 sH0[BT * SH_STR];
    __shared__ __align__(16) _Float16 sH1[BT * SH_STR];
    __shared__ int sTok[BT * T_LEN];

    const int tid   = threadIdx.x;
    const int lane  = tid & 31;
    const int wave  = tid >> 5;
    const int bBase = blockIdx.x * BT;

    // zero hidden state AND sX (pad columns of sX stay zero forever)
    for (int i = tid; i < BT * SH_STR; i += 256) {
        sH0[i] = (_Float16)0.0f;
        sH1[i] = (_Float16)0.0f;
    }
    for (int i = tid; i < BT * SX_STR; i += 256) sX[i] = (_Float16)0.0f;
    // preload this block's tokens (fully coalesced, done once)
    for (int i = tid; i < BT * T_LEN; i += 256) sTok[i] = tokens[bBase * T_LEN + i];

    // per-lane bias scalars for this wave's 4 column tiles
    float bv0[NT_PW], bv1[NT_PW];
#pragma unroll
    for (int j = 0; j < NT_PW; ++j) {
        const int col = (wave * NT_PW + j) * 16 + (lane & 15);
        bv0[j] = b0[col];
        bv1[j] = b1[col];
    }
    __syncthreads();

    const int mrow = (lane >> 4) * 8;   // C-fragment row base per lane half
    const int ncol = lane & 15;
    const int gr   = tid >> 3;          // gather: 32 rows x 8 threads
    const int gs   = tid & 7;

    for (int t = 0; t < T_LEN; ++t) {
        // ---- embedding gather: [BT][E_DIM] fp32 -> [BT][0..E_DIM) f16 (pad pre-zeroed)
        {
            const int tok = sTok[gr * T_LEN + t];
            const float* erow = emb + (long)tok * E_DIM;
#pragma unroll
            for (int i = 0; i < 4; ++i) {
                const int c4 = (gs + 8 * i) * 4;    // 0,4,...,124
                if (c4 < E_DIM) {                   // 25 valid 4-wide chunks
                    const float4 v = *(const float4*)(erow + c4);
                    v4h h;
                    h[0] = (_Float16)v.x; h[1] = (_Float16)v.y;
                    h[2] = (_Float16)v.z; h[3] = (_Float16)v.w;
                    *(v4h*)(sX + gr * SX_STR + c4) = h;
                }
            }
        }
        __syncthreads();

        // ---- layer 0: a0 = x @ W0p + h0 @ U0 + b0 ----
        v8f acc[2][NT_PW];
#pragma unroll
        for (int j = 0; j < NT_PW; ++j) {
#pragma unroll
            for (int r = 0; r < 8; ++r) { acc[0][j][r] = bv0[j]; acc[1][j][r] = bv0[j]; }
        }
#pragma unroll
        for (int kt = 0; kt < KT_E; ++kt) {
            v16h a0 = load_a(sX,               SX_STR, lane, kt * 32);
            v16h a1 = load_a(sX + 16 * SX_STR, SX_STR, lane, kt * 32);
#pragma unroll
            for (int j = 0; j < NT_PW; ++j) {
                const int nt = wave * NT_PW + j;
                v16h b = *(const v16h*)(wsW0 + (((nt * KT_E + kt) * 32 + lane) << 4));
                acc[0][j] = wmma16(a0, b, acc[0][j]);
                acc[1][j] = wmma16(a1, b, acc[1][j]);
            }
        }
        for (int kt = 0; kt < KT_U; ++kt) {
            v16h a0 = load_a(sH0,               SH_STR, lane, kt * 32);
            v16h a1 = load_a(sH0 + 16 * SH_STR, SH_STR, lane, kt * 32);
#pragma unroll
            for (int j = 0; j < NT_PW; ++j) {
                const int nt = wave * NT_PW + j;
                v16h b = *(const v16h*)(wsU0 + (((nt * KT_U + kt) * 32 + lane) << 4));
                acc[0][j] = wmma16(a0, b, acc[0][j]);
                acc[1][j] = wmma16(a1, b, acc[1][j]);
            }
        }
        __syncthreads();   // everyone done reading old h0

        // tanh -> new h0 (C-fragment: VGPR r => row mrow+r, col = lane&15)
#pragma unroll
        for (int m = 0; m < 2; ++m)
#pragma unroll
            for (int j = 0; j < NT_PW; ++j) {
                const int colBase = (wave * NT_PW + j) * 16 + ncol;
#pragma unroll
                for (int r = 0; r < 8; ++r)
                    sH0[(m * 16 + mrow + r) * SH_STR + colBase] =
                        (_Float16)fast_tanh(acc[m][j][r]);
            }
        __syncthreads();

        // ---- layer 1: a1 = h0_new @ W1 + h1 @ U1 + b1 ----
#pragma unroll
        for (int j = 0; j < NT_PW; ++j) {
#pragma unroll
            for (int r = 0; r < 8; ++r) { acc[0][j][r] = bv1[j]; acc[1][j][r] = bv1[j]; }
        }
        for (int kt = 0; kt < KT_U; ++kt) {
            v16h a0 = load_a(sH0,               SH_STR, lane, kt * 32);
            v16h a1 = load_a(sH0 + 16 * SH_STR, SH_STR, lane, kt * 32);
            v16h c0 = load_a(sH1,               SH_STR, lane, kt * 32);
            v16h c1 = load_a(sH1 + 16 * SH_STR, SH_STR, lane, kt * 32);
#pragma unroll
            for (int j = 0; j < NT_PW; ++j) {
                const int nt = wave * NT_PW + j;
                v16h bw = *(const v16h*)(wsW1 + (((nt * KT_U + kt) * 32 + lane) << 4));
                v16h bu = *(const v16h*)(wsU1 + (((nt * KT_U + kt) * 32 + lane) << 4));
                acc[0][j] = wmma16(a0, bw, acc[0][j]);
                acc[1][j] = wmma16(a1, bw, acc[1][j]);
                acc[0][j] = wmma16(c0, bu, acc[0][j]);
                acc[1][j] = wmma16(c1, bu, acc[1][j]);
            }
        }
        __syncthreads();   // done reading old h1
#pragma unroll
        for (int m = 0; m < 2; ++m)
#pragma unroll
            for (int j = 0; j < NT_PW; ++j) {
                const int colBase = (wave * NT_PW + j) * 16 + ncol;
#pragma unroll
                for (int r = 0; r < 8; ++r)
                    sH1[(m * 16 + mrow + r) * SH_STR + colBase] =
                        (_Float16)fast_tanh(acc[m][j][r]);
            }
        __syncthreads();
    }

    // ---- output head: out[b] = sigmoid(h1[b] . Wout + bout) ----
    {
        const int r   = tid >> 3;         // 32 rows x 8 threads
        const int seg = tid & 7;
        float s = 0.0f;
        for (int k = seg * 64; k < seg * 64 + 64; ++k)
            s += (float)sH1[r * SH_STR + k] * Wout[k];
        s += __shfl_xor(s, 1, 32);
        s += __shfl_xor(s, 2, 32);
        s += __shfl_xor(s, 4, 32);
        if (seg == 0) {
            const float z = s + bout[0];
            out[bBase + r] = 1.0f / (1.0f + __expf(-z));
        }
    }
}

extern "C" void kernel_launch(void* const* d_in, const int* in_sizes, int n_in,
                              void* d_out, int out_size, void* d_ws, size_t ws_size,
                              hipStream_t stream)
{
    const int*   tokens = (const int*)  d_in[0];
    const float* emb    = (const float*)d_in[1];
    const float* W0     = (const float*)d_in[2];
    const float* U0     = (const float*)d_in[3];
    const float* b0     = (const float*)d_in[4];
    const float* W1     = (const float*)d_in[5];
    const float* U1     = (const float*)d_in[6];
    const float* b1     = (const float*)d_in[7];
    const float* Wout   = (const float*)d_in[8];
    const float* bout   = (const float*)d_in[9];
    float*       out    = (float*)d_out;

    _Float16* ws = (_Float16*)d_ws;
    _Float16* wsW0 = ws;
    _Float16* wsU0 = wsW0 + EPAD * U_DIM;
    _Float16* wsW1 = wsU0 + U_DIM * U_DIM;
    _Float16* wsU1 = wsW1 + U_DIM * U_DIM;

    prep_weights<<<832, 1024, 0, stream>>>(W0, U0, W1, U1, ws);
    rnn_main<<<B_TOT / BT, 256, 0, stream>>>(tokens, emb, wsW0, wsU0, wsW1, wsU1,
                                             b0, b1, Wout, bout, out);
}
